// PairformerModule_50173807952270
// MI455X (gfx1250) — matrix operand order repaired and checked by reading, not compiled, not run on hardware
//
#include <hip/hip_runtime.h>
#include <stdint.h>

// ---------------------------------------------------------------------------
// Pairformer (AlphaFold3-style) for MI455X / gfx1250, wave32 + WMMA bf16.
// All large GEMMs -> v_wmma_f32_16x16x32_bf16 (bf16 in, fp32 accum), with
// 2x4 register blocking per wave (32x64 macro-tile) for operand reuse.
// LN / softmax / gating -> fp32 VALU kernels.
// ---------------------------------------------------------------------------

#define N_   192
#define NN_  (192 * 192)
#define TZ_  128
#define TS_  384
#define INF_ 1e9f

typedef __attribute__((ext_vector_type(16))) __bf16 v16bf;
typedef __attribute__((ext_vector_type(8)))  float  v8f;

struct __attribute__((aligned(16))) U128 { unsigned int x[4]; };
union FragU { v16bf v; U128 q[2]; };

__device__ __forceinline__ unsigned short f2bf(float f) {
  unsigned int u = __float_as_uint(f);
  unsigned int r = (u + 0x7fffu + ((u >> 16) & 1u)) >> 16;
  return (unsigned short)r;
}
__device__ __forceinline__ float sigm(float x) { return 1.f / (1.f + __expf(-x)); }

__device__ __forceinline__ v16bf ld_frag_a(const unsigned short* arow, int k, int g) {
  FragU a;
  a.q[0] = *(const U128*)(arow + k + 8 * g);       // K = 8g..8g+7
  a.q[1] = *(const U128*)(arow + k + 16 + 8 * g);  // K = 16+8g..+7
  return a.v;
}
__device__ __forceinline__ v16bf ld_frag_b(const unsigned short* brow, int k, int g) {
  FragU b;
  b.q[0] = *(const U128*)(brow + k + 8 * g);       // K = 8g..8g+7
  b.q[1] = *(const U128*)(brow + k + 16 + 8 * g);  // K = 16+8g..+7
  return b.v;
}

// --------------------------- WMMA GEMM (1x1 fallback) ----------------------
// C[M,Nc] = A[M,K](bf16,row-major) * BT[Nc,K](bf16,row-major)^T  (+ bias[n])
__global__ void k_gemm(const unsigned short* __restrict__ A,
                       const unsigned short* __restrict__ BT,
                       float* __restrict__ C, const float* __restrict__ bias,
                       int M, int Nc, int K,
                       long long sA, long long sB, long long sC) {
  int bat = blockIdx.y;
  A  += (size_t)bat * sA;
  BT += (size_t)bat * sB;
  C  += (size_t)bat * sC;
  int wave   = blockIdx.x * (blockDim.x >> 5) + (threadIdx.x >> 5);
  int tilesN = Nc >> 4;
  int tiles  = (M >> 4) * tilesN;
  if (wave >= tiles) return;
  int tm = wave / tilesN, tn = wave - tm * tilesN;
  int lane = threadIdx.x & 31;
  int l15 = lane & 15, g = lane >> 4;
  const unsigned short* arow = A  + (size_t)(tm * 16 + l15) * K;
  const unsigned short* brow = BT + (size_t)(tn * 16 + l15) * K;
  v8f acc = {0.f, 0.f, 0.f, 0.f, 0.f, 0.f, 0.f, 0.f};
  for (int k = 0; k < K; k += 32) {
    acc = __builtin_amdgcn_wmma_f32_16x16x32_bf16(false, ld_frag_a(arow, k, g),
                                                  false, ld_frag_b(brow, k, g),
                                                  (short)0, acc, false, false);
  }
  int n = tn * 16 + l15;
  float bv = bias ? bias[n] : 0.f;
#pragma unroll
  for (int r = 0; r < 8; ++r) {
    int m = tm * 16 + r + 8 * g;
    C[(size_t)m * Nc + n] = acc[r] + bv;
  }
}

// ------------------- WMMA GEMM, 2x4 register-blocked -----------------------
// Each wave: 32x64 macro-tile = 8 WMMA tiles; per K-step 2 A-frags + 4 B-frags
// feed 8 v_wmma (1.5 b128 loads per WMMA).  Requires M%32==0, Nc%64==0.
__global__ void k_gemm_b24(const unsigned short* __restrict__ A,
                           const unsigned short* __restrict__ BT,
                           float* __restrict__ C, const float* __restrict__ bias,
                           int M, int Nc, int K,
                           long long sA, long long sB, long long sC) {
  int bat = blockIdx.y;
  A  += (size_t)bat * sA;
  BT += (size_t)bat * sB;
  C  += (size_t)bat * sC;
  int wave   = blockIdx.x * (blockDim.x >> 5) + (threadIdx.x >> 5);
  int tilesN = Nc >> 6;                // 64-col macro tiles
  int tiles  = (M >> 5) * tilesN;      // 32-row macro tiles
  if (wave >= tiles) return;
  int tm = wave / tilesN, tn = wave - tm * tilesN;
  int lane = threadIdx.x & 31;
  int l15 = lane & 15, g = lane >> 4;
  const unsigned short* ar0 = A + (size_t)(tm * 32 + l15) * K;
  const unsigned short* ar1 = ar0 + (size_t)16 * K;
  const unsigned short* br0 = BT + (size_t)(tn * 64 + l15) * K;
  const unsigned short* br1 = br0 + (size_t)16 * K;
  const unsigned short* br2 = br0 + (size_t)32 * K;
  const unsigned short* br3 = br0 + (size_t)48 * K;
  v8f acc[2][4];
#pragma unroll
  for (int mi = 0; mi < 2; ++mi)
#pragma unroll
    for (int ni = 0; ni < 4; ++ni)
      acc[mi][ni] = (v8f){0.f, 0.f, 0.f, 0.f, 0.f, 0.f, 0.f, 0.f};
  for (int k = 0; k < K; k += 32) {
    v16bf a0 = ld_frag_a(ar0, k, g);
    v16bf a1 = ld_frag_a(ar1, k, g);
    v16bf b0 = ld_frag_b(br0, k, g);
    v16bf b1 = ld_frag_b(br1, k, g);
    v16bf b2 = ld_frag_b(br2, k, g);
    v16bf b3 = ld_frag_b(br3, k, g);
    acc[0][0] = __builtin_amdgcn_wmma_f32_16x16x32_bf16(false, a0, false, b0, (short)0, acc[0][0], false, false);
    acc[0][1] = __builtin_amdgcn_wmma_f32_16x16x32_bf16(false, a0, false, b1, (short)0, acc[0][1], false, false);
    acc[0][2] = __builtin_amdgcn_wmma_f32_16x16x32_bf16(false, a0, false, b2, (short)0, acc[0][2], false, false);
    acc[0][3] = __builtin_amdgcn_wmma_f32_16x16x32_bf16(false, a0, false, b3, (short)0, acc[0][3], false, false);
    acc[1][0] = __builtin_amdgcn_wmma_f32_16x16x32_bf16(false, a1, false, b0, (short)0, acc[1][0], false, false);
    acc[1][1] = __builtin_amdgcn_wmma_f32_16x16x32_bf16(false, a1, false, b1, (short)0, acc[1][1], false, false);
    acc[1][2] = __builtin_amdgcn_wmma_f32_16x16x32_bf16(false, a1, false, b2, (short)0, acc[1][2], false, false);
    acc[1][3] = __builtin_amdgcn_wmma_f32_16x16x32_bf16(false, a1, false, b3, (short)0, acc[1][3], false, false);
  }
#pragma unroll
  for (int mi = 0; mi < 2; ++mi) {
#pragma unroll
    for (int ni = 0; ni < 4; ++ni) {
      int n = tn * 64 + ni * 16 + l15;
      float bv = bias ? bias[n] : 0.f;
#pragma unroll
      for (int r = 0; r < 8; ++r) {
        int m = tm * 32 + mi * 16 + r + 8 * g;
        C[(size_t)m * Nc + n] = acc[mi][ni][r] + bv;
      }
    }
  }
}

// Fused transition core: H[M,Nc](bf16) = silu(A@BT1^T) * (A@BT2^T)
// 2x M-blocking: 8 b128 loads -> 4 WMMAs per K-step.  Requires M%32==0.
__global__ void k_gemm_dual_silu(const unsigned short* __restrict__ A,
                                 const unsigned short* __restrict__ BT1,
                                 const unsigned short* __restrict__ BT2,
                                 unsigned short* __restrict__ H,
                                 int M, int Nc, int K) {
  int wave   = blockIdx.x * (blockDim.x >> 5) + (threadIdx.x >> 5);
  int tilesN = Nc >> 4;
  int tiles  = (M >> 5) * tilesN;
  if (wave >= tiles) return;
  int tm = wave / tilesN, tn = wave - tm * tilesN;
  int lane = threadIdx.x & 31;
  int l15 = lane & 15, g = lane >> 4;
  const unsigned short* ar0 = A + (size_t)(tm * 32 + l15) * K;
  const unsigned short* ar1 = ar0 + (size_t)16 * K;
  const unsigned short* b1r = BT1 + (size_t)(tn * 16 + l15) * K;
  const unsigned short* b2r = BT2 + (size_t)(tn * 16 + l15) * K;
  v8f a1c[2], a2c[2];
#pragma unroll
  for (int mi = 0; mi < 2; ++mi) {
    a1c[mi] = (v8f){0.f, 0.f, 0.f, 0.f, 0.f, 0.f, 0.f, 0.f};
    a2c[mi] = (v8f){0.f, 0.f, 0.f, 0.f, 0.f, 0.f, 0.f, 0.f};
  }
  for (int k = 0; k < K; k += 32) {
    v16bf a0 = ld_frag_a(ar0, k, g);
    v16bf a1 = ld_frag_a(ar1, k, g);
    v16bf b1 = ld_frag_b(b1r, k, g);
    v16bf b2 = ld_frag_b(b2r, k, g);
    a1c[0] = __builtin_amdgcn_wmma_f32_16x16x32_bf16(false, a0, false, b1, (short)0, a1c[0], false, false);
    a2c[0] = __builtin_amdgcn_wmma_f32_16x16x32_bf16(false, a0, false, b2, (short)0, a2c[0], false, false);
    a1c[1] = __builtin_amdgcn_wmma_f32_16x16x32_bf16(false, a1, false, b1, (short)0, a1c[1], false, false);
    a2c[1] = __builtin_amdgcn_wmma_f32_16x16x32_bf16(false, a1, false, b2, (short)0, a2c[1], false, false);
  }
  int n = tn * 16 + l15;
#pragma unroll
  for (int mi = 0; mi < 2; ++mi) {
#pragma unroll
    for (int r = 0; r < 8; ++r) {
      int m = tm * 32 + mi * 16 + r + 8 * g;
      float c1 = a1c[mi][r];
      H[(size_t)m * Nc + n] = f2bf(c1 * sigm(c1) * a2c[mi][r]);
    }
  }
}

// --------------------------- LayerNorm -------------------------------------
__global__ void k_ln(const float* __restrict__ X, const float* __restrict__ g,
                     const float* __restrict__ b, float* __restrict__ Yf,
                     unsigned short* __restrict__ Yb, int D, int transN) {
  __shared__ float red[128];
  int r = blockIdx.x, src = r;
  if (transN) { int a = r / transN, c = r - a * transN; src = c * transN + a; }
  const float* x = X + (size_t)src * D;
  int tid = threadIdx.x;
  float s = 0.f;
  for (int d = tid; d < D; d += 128) s += x[d];
  red[tid] = s; __syncthreads();
  for (int st = 64; st > 0; st >>= 1) { if (tid < st) red[tid] += red[tid + st]; __syncthreads(); }
  float mean = red[0] / D;
  __syncthreads();
  float v = 0.f;
  for (int d = tid; d < D; d += 128) { float t = x[d] - mean; v += t * t; }
  red[tid] = v; __syncthreads();
  for (int st = 64; st > 0; st >>= 1) { if (tid < st) red[tid] += red[tid + st]; __syncthreads(); }
  float rstd = rsqrtf(red[0] / D + 1e-5f);
  for (int d = tid; d < D; d += 128) {
    float y = (x[d] - mean) * rstd * g[d] + b[d];
    if (Yf) Yf[(size_t)r * D + d] = y;
    if (Yb) Yb[(size_t)r * D + d] = f2bf(y);
  }
}

// LN over channel dim of channel-major tensor T[128][rows] -> bf16 [rows,128]
__global__ void k_ln_chan(const float* __restrict__ T, const float* __restrict__ g,
                          const float* __restrict__ b, unsigned short* __restrict__ Yb,
                          int rows) {
  __shared__ float red[128];
  int r = blockIdx.x, d = threadIdx.x;
  float x = T[(size_t)d * rows + r];
  red[d] = x; __syncthreads();
  for (int st = 64; st > 0; st >>= 1) { if (d < st) red[d] += red[d + st]; __syncthreads(); }
  float mean = red[0] / 128.f;
  __syncthreads();
  float t = x - mean;
  red[d] = t * t; __syncthreads();
  for (int st = 64; st > 0; st >>= 1) { if (d < st) red[d] += red[d + st]; __syncthreads(); }
  float rstd = rsqrtf(red[0] / 128.f + 1e-5f);
  Yb[(size_t)r * 128 + d] = f2bf(t * rstd * g[d] + b[d]);
}

// ---------------- tri_mul helpers ------------------------------------------
__global__ void k_pack_ab(const float* __restrict__ HP, const float* __restrict__ HG,
                          const float* __restrict__ pm, unsigned short* __restrict__ At,
                          unsigned short* __restrict__ Bt, int incoming) {
  size_t idx = (size_t)blockIdx.x * 256 + threadIdx.x;
  if (idx >= (size_t)NN_ * 256) return;
  int d2 = (int)(idx & 255);
  size_t rc = idx >> 8;
  int c = (int)(rc % N_), r = (int)(rc / N_);
  float v = HP[idx] * sigm(HG[idx]) * pm[rc];
  int d = d2 & 127;
  int ri = incoming ? c : r;
  int ci = incoming ? r : c;
  unsigned short* dst = (d2 < 128) ? At : Bt;
  dst[(size_t)d * NN_ + (size_t)ri * N_ + ci] = f2bf(v);
}

__global__ void k_gate_add(float* __restrict__ Z, const float* __restrict__ P,
                           const float* __restrict__ G, size_t cnt) {
  size_t i = (size_t)blockIdx.x * 256 + threadIdx.x;
  if (i < cnt) Z[i] += P[i] * sigm(G[i]);
}
__global__ void k_gate_bf16(unsigned short* __restrict__ H, const float* __restrict__ O,
                            const float* __restrict__ G, size_t cnt) {
  size_t i = (size_t)blockIdx.x * 256 + threadIdx.x;
  if (i < cnt) H[i] = f2bf(O[i] * sigm(G[i]));
}
__global__ void k_add(float* __restrict__ D, const float* __restrict__ S, size_t cnt) {
  size_t i = (size_t)blockIdx.x * 256 + threadIdx.x;
  if (i < cnt) D[i] += S[i];
}
__global__ void k_add_tr(float* __restrict__ Z, const float* __restrict__ X) {
  size_t idx = (size_t)blockIdx.x * 256 + threadIdx.x;
  if (idx >= (size_t)NN_ * 128) return;
  int d = (int)(idx & 127);
  size_t rc = idx >> 7;
  int j = (int)(rc % N_), i = (int)(rc / N_);
  Z[((size_t)j * N_ + i) * 128 + d] += X[idx];
}

// tb[r,h] = dot(XF[r,:128], lin_b[:,h])
__global__ void k_linb(const float* __restrict__ XF, const float* __restrict__ W,
                       float* __restrict__ TB) {
  size_t idx = (size_t)blockIdx.x * 256 + threadIdx.x;
  if (idx >= (size_t)NN_ * 4) return;
  int h = (int)(idx & 3);
  size_t r = idx >> 2;
  const float* x = XF + r * 128;
  float s = 0.f;
  for (int d = 0; d < 128; ++d) s += x[d] * W[d * 4 + h];
  TB[idx] = s;
}

// ---------------- attention (online softmax, fp32 VALU) --------------------
__global__ void __launch_bounds__(192)
k_tri_attn(const float* __restrict__ Q, const float* __restrict__ Kp,
           const float* __restrict__ V, const float* __restrict__ TB,
           const float* __restrict__ pm, float* __restrict__ O, int trans) {
  int i = blockIdx.x, h = blockIdx.y, q = threadIdx.x;
  const float scale = 0.1767766952966369f;  // 32^-0.5
  const float* qp = Q + ((size_t)(i * N_ + q)) * 128 + h * 32;
  float qr[32];
#pragma unroll
  for (int d = 0; d < 32; ++d) qr[d] = qp[d] * scale;
  float m = -1e30f, l = 0.f, o[32];
#pragma unroll
  for (int d = 0; d < 32; ++d) o[d] = 0.f;
  for (int k = 0; k < N_; ++k) {
    const float* kp = Kp + ((size_t)(i * N_ + k)) * 128 + h * 32;
    float s = 0.f;
#pragma unroll
    for (int d = 0; d < 32; ++d) s += qr[d] * kp[d];
    float pmv = trans ? pm[(size_t)k * N_ + i] : pm[(size_t)i * N_ + k];
    s += INF_ * (pmv - 1.f);
    s += TB[((size_t)q * N_ + k) * 4 + h];
    float mn = fmaxf(m, s);
    float cor = __expf(m - mn), e = __expf(s - mn);
    const float* vp = V + ((size_t)(i * N_ + k)) * 128 + h * 32;
    l = l * cor + e;
#pragma unroll
    for (int d = 0; d < 32; ++d) o[d] = o[d] * cor + e * vp[d];
    m = mn;
  }
  float inv = 1.f / l;
  float* op = O + ((size_t)(i * N_ + q)) * 128 + h * 32;
#pragma unroll
  for (int d = 0; d < 32; ++d) op[d] = o[d] * inv;
}

__global__ void __launch_bounds__(192)
k_apb_attn(const float* __restrict__ Q, const float* __restrict__ K,
           const float* __restrict__ V, const float* __restrict__ BZ,
           const float* __restrict__ mask, float* __restrict__ O) {
  int h = blockIdx.x, q = threadIdx.x;
  const float scale = 0.2041241452319315f;  // 24^-0.5
  const float* qp = Q + (size_t)q * TS_ + h * 24;
  float qr[24];
#pragma unroll
  for (int d = 0; d < 24; ++d) qr[d] = qp[d] * scale;
  float m = -1e30f, l = 0.f, o[24];
#pragma unroll
  for (int d = 0; d < 24; ++d) o[d] = 0.f;
  for (int j = 0; j < N_; ++j) {
    const float* kp = K + (size_t)j * TS_ + h * 24;
    float s = 0.f;
#pragma unroll
    for (int d = 0; d < 24; ++d) s += qr[d] * kp[d];
    s += BZ[((size_t)q * N_ + j) * 16 + h];
    s += (mask[j] - 1.f) * INF_;
    float mn = fmaxf(m, s);
    float cor = __expf(m - mn), e = __expf(s - mn);
    const float* vp = V + (size_t)j * TS_ + h * 24;
    l = l * cor + e;
#pragma unroll
    for (int d = 0; d < 24; ++d) o[d] = o[d] * cor + e * vp[d];
    m = mn;
  }
  float inv = 1.f / l;
  float* op = O + (size_t)q * TS_ + h * 24;
#pragma unroll
  for (int d = 0; d < 24; ++d) op[d] = o[d] * inv;
}

// Weight fp32 [Kin,Nout] -> bf16 transposed [Nout,Kin]
__global__ void k_w_to_bf16t(const float* __restrict__ W, unsigned short* __restrict__ WT,
                             int Kin, int Nout) {
  size_t idx = (size_t)blockIdx.x * 256 + threadIdx.x;
  if (idx >= (size_t)Kin * Nout) return;
  int n = (int)(idx % Nout);
  size_t k = idx / Nout;
  WT[(size_t)n * Kin + k] = f2bf(W[idx]);
}

// ---------------------------------------------------------------------------
extern "C" void kernel_launch(void* const* d_in, const int* in_sizes, int n_in,
                              void* d_out, int out_size, void* d_ws, size_t ws_size,
                              hipStream_t stream) {
  (void)in_sizes; (void)n_in; (void)out_size; (void)ws_size;
  const float* s_in  = (const float*)d_in[0];
  const float* z_in  = (const float*)d_in[1];
  const float* maskv = (const float*)d_in[2];
  const float* pmask = (const float*)d_in[3];
  // params flattened in setup_inputs() insertion order:
  //   per block (57 leaves): tmo[8], tmi[8], tas[10], tae[10], apb[11], tz[5], ts[5]
  auto P = [&](int blk, int idx) -> const float* {
    return (const float*)d_in[4 + blk * 57 + idx];
  };

  char* base = (char*)d_ws;
  size_t off = 0;
  auto alloc = [&](size_t bytes) -> char* {
    char* p = base + off;
    off += (bytes + 255) & ~(size_t)255;
    return p;
  };

  float* Z  = (float*)alloc((size_t)NN_ * TZ_ * 4);
  float* S  = (float*)alloc((size_t)N_ * TS_ * 4);
  float* XF = (float*)alloc((size_t)NN_ * TZ_ * 4);
  unsigned short* XB = (unsigned short*)alloc((size_t)NN_ * TZ_ * 2);
  float* G1 = (float*)alloc((size_t)NN_ * 256 * 4);
  float* G2 = (float*)alloc((size_t)NN_ * 256 * 4);
  float* G3 = (float*)alloc((size_t)NN_ * TZ_ * 4);
  float* G4 = (float*)alloc((size_t)NN_ * TZ_ * 4);
  unsigned short* HBIG = (unsigned short*)alloc((size_t)NN_ * 512 * 2);
  unsigned short* HB1 = HBIG;
  unsigned short* HB2 = HBIG + (size_t)NN_ * 128;
  unsigned short* HB3 = HBIG + (size_t)NN_ * 256;
  float* SXF = (float*)alloc((size_t)N_ * TS_ * 4);
  unsigned short* SXB = (unsigned short*)alloc((size_t)N_ * TS_ * 2);
  float* Qb = (float*)alloc((size_t)N_ * TS_ * 4);
  float* Kb = (float*)alloc((size_t)N_ * TS_ * 4);
  float* Vb = (float*)alloc((size_t)N_ * TS_ * 4);
  float* Gb = (float*)alloc((size_t)N_ * TS_ * 4);
  float* Ob = (float*)alloc((size_t)N_ * TS_ * 4);
  float* Pb = (float*)alloc((size_t)N_ * TS_ * 4);
  unsigned short* SG16 = (unsigned short*)alloc((size_t)N_ * TS_ * 2);
  unsigned short* SH16 = (unsigned short*)alloc((size_t)N_ * 1536 * 2);

  hipMemcpyAsync(Z, z_in, (size_t)NN_ * TZ_ * 4, hipMemcpyDeviceToDevice, stream);
  hipMemcpyAsync(S, s_in, (size_t)N_ * TS_ * 4, hipMemcpyDeviceToDevice, stream);

  auto cvtw = [&](const float* W, int Kin, int Nout) -> unsigned short* {
    unsigned short* WT = (unsigned short*)alloc((size_t)Kin * Nout * 2);
    size_t tot = (size_t)Kin * Nout;
    k_w_to_bf16t<<<dim3((unsigned)((tot + 255) / 256)), 256, 0, stream>>>(W, WT, Kin, Nout);
    return WT;
  };
  auto gemm = [&](const unsigned short* A, const unsigned short* BT, float* C,
                  const float* bias, int M, int Nc, int K, int batch,
                  long long sA, long long sB, long long sC) {
    if ((M & 31) == 0 && (Nc & 63) == 0) {
      int tiles = (M / 32) * (Nc / 64);
      k_gemm_b24<<<dim3((unsigned)((tiles + 7) / 8), (unsigned)batch), 256, 0, stream>>>(
          A, BT, C, bias, M, Nc, K, sA, sB, sC);
    } else {
      int tiles = (M / 16) * (Nc / 16);
      k_gemm<<<dim3((unsigned)((tiles + 7) / 8), (unsigned)batch), 256, 0, stream>>>(
          A, BT, C, bias, M, Nc, K, sA, sB, sC);
    }
  };
  auto dual = [&](const unsigned short* A, const unsigned short* B1,
                  const unsigned short* B2, unsigned short* H, int M, int Nc, int K) {
    int tiles = (M / 32) * (Nc / 16);
    k_gemm_dual_silu<<<dim3((unsigned)((tiles + 7) / 8)), 256, 0, stream>>>(A, B1, B2, H, M, Nc, K);
  };
  auto ew = [&](size_t cnt) { return dim3((unsigned)((cnt + 255) / 256)); };

  for (int blk = 0; blk < 2; ++blk) {
    // --- weight conversion (bf16, transposed) ---
    unsigned short *tm_pinT[2], *tm_ginT[2], *tm_poutT[2], *tm_goutT[2];
    for (int t = 0; t < 2; ++t) {
      int b0 = t * 8;
      tm_pinT[t]  = cvtw(P(blk, b0 + 2), TZ_, 256);
      tm_ginT[t]  = cvtw(P(blk, b0 + 3), TZ_, 256);
      tm_poutT[t] = cvtw(P(blk, b0 + 6), TZ_, TZ_);
      tm_goutT[t] = cvtw(P(blk, b0 + 7), TZ_, TZ_);
    }
    unsigned short *ta_wqT[2], *ta_wkT[2], *ta_wvT[2], *ta_wgT[2], *ta_woT[2];
    for (int t = 0; t < 2; ++t) {
      int b0 = 16 + t * 10;
      ta_wqT[t] = cvtw(P(blk, b0 + 3), TZ_, 128);
      ta_wkT[t] = cvtw(P(blk, b0 + 4), TZ_, 128);
      ta_wvT[t] = cvtw(P(blk, b0 + 5), TZ_, 128);
      ta_wgT[t] = cvtw(P(blk, b0 + 6), TZ_, 128);
      ta_woT[t] = cvtw(P(blk, b0 + 8), 128, TZ_);
    }
    unsigned short* ap_wqT = cvtw(P(blk, 36 + 2), TS_, TS_);
    unsigned short* ap_wkT = cvtw(P(blk, 36 + 4), TS_, TS_);
    unsigned short* ap_wvT = cvtw(P(blk, 36 + 5), TS_, TS_);
    unsigned short* ap_wzT = cvtw(P(blk, 36 + 8), TZ_, 16);
    unsigned short* ap_wgT = cvtw(P(blk, 36 + 9), TS_, TS_);
    unsigned short* ap_woT = cvtw(P(blk, 36 + 10), TS_, TS_);
    unsigned short* tz_w1T = cvtw(P(blk, 47 + 2), TZ_, 512);
    unsigned short* tz_w2T = cvtw(P(blk, 47 + 3), TZ_, 512);
    unsigned short* tz_w3T = cvtw(P(blk, 47 + 4), 512, TZ_);
    unsigned short* ts_w1T = cvtw(P(blk, 52 + 2), TS_, 1536);
    unsigned short* ts_w2T = cvtw(P(blk, 52 + 3), TS_, 1536);
    unsigned short* ts_w3T = cvtw(P(blk, 52 + 4), 1536, TS_);

    // --- triangle multiplication: outgoing (t=0), incoming (t=1) ---
    for (int t = 0; t < 2; ++t) {
      int b0 = t * 8;
      k_ln<<<NN_, 128, 0, stream>>>(Z, P(blk, b0 + 0), P(blk, b0 + 1), nullptr, XB, TZ_, 0);
      gemm(XB, tm_pinT[t], G1, nullptr, NN_, 256, TZ_, 1, 0, 0, 0);
      gemm(XB, tm_ginT[t], G2, nullptr, NN_, 256, TZ_, 1, 0, 0, 0);
      k_pack_ab<<<ew((size_t)NN_ * 256), 256, 0, stream>>>(G1, G2, pmask, HB1, HB2, t);
      gemm(HB1, HB2, G3, nullptr, N_, N_, N_, TZ_, NN_, NN_, NN_);  // 128 batched GEMMs
      k_ln_chan<<<NN_, 128, 0, stream>>>(G3, P(blk, b0 + 4), P(blk, b0 + 5), HB3, NN_);
      gemm(XB, tm_goutT[t], G4, nullptr, NN_, TZ_, TZ_, 1, 0, 0, 0);
      gemm(HB3, tm_poutT[t], G1, nullptr, NN_, TZ_, TZ_, 1, 0, 0, 0);
      k_gate_add<<<ew((size_t)NN_ * TZ_), 256, 0, stream>>>(Z, G1, G4, (size_t)NN_ * TZ_);
    }

    // --- triangle attention: starting (t=0), ending (t=1, transposed frame) ---
    for (int t = 0; t < 2; ++t) {
      int b0 = 16 + t * 10;
      k_ln<<<NN_, 128, 0, stream>>>(Z, P(blk, b0 + 0), P(blk, b0 + 1), XF, XB, TZ_, t ? N_ : 0);
      gemm(XB, ta_wqT[t], G1, nullptr, NN_, 128, TZ_, 1, 0, 0, 0);
      gemm(XB, ta_wkT[t], G1 + (size_t)NN_ * 128, nullptr, NN_, 128, TZ_, 1, 0, 0, 0);
      gemm(XB, ta_wvT[t], G2, nullptr, NN_, 128, TZ_, 1, 0, 0, 0);
      gemm(XB, ta_wgT[t], G2 + (size_t)NN_ * 128, P(blk, b0 + 7), NN_, 128, TZ_, 1, 0, 0, 0);
      k_linb<<<ew((size_t)NN_ * 4), 256, 0, stream>>>(XF, P(blk, b0 + 2), G3);
      k_tri_attn<<<dim3(N_, 4), 192, 0, stream>>>(G1, G1 + (size_t)NN_ * 128, G2, G3,
                                                  pmask, G4, t);
      k_gate_bf16<<<ew((size_t)NN_ * 128), 256, 0, stream>>>(HB1, G4, G2 + (size_t)NN_ * 128,
                                                             (size_t)NN_ * 128);
      gemm(HB1, ta_woT[t], G3, P(blk, b0 + 9), NN_, TZ_, 128, 1, 0, 0, 0);
      if (t) k_add_tr<<<ew((size_t)NN_ * TZ_), 256, 0, stream>>>(Z, G3);
      else   k_add<<<ew((size_t)NN_ * TZ_), 256, 0, stream>>>(Z, G3, (size_t)NN_ * TZ_);
    }

    // --- pair transition ---
    k_ln<<<NN_, 128, 0, stream>>>(Z, P(blk, 47 + 0), P(blk, 47 + 1), nullptr, XB, TZ_, 0);
    dual(XB, tz_w1T, tz_w2T, HBIG, NN_, 512, TZ_);
    gemm(HBIG, tz_w3T, G3, nullptr, NN_, TZ_, 512, 1, 0, 0, 0);
    k_add<<<ew((size_t)NN_ * TZ_), 256, 0, stream>>>(Z, G3, (size_t)NN_ * TZ_);

    // --- attention with pair bias (sequence track) ---
    k_ln<<<N_, 128, 0, stream>>>(S, P(blk, 36 + 0), P(blk, 36 + 1), SXF, SXB, TS_, 0);
    gemm(SXB, ap_wqT, Qb, P(blk, 36 + 3), N_, TS_, TS_, 1, 0, 0, 0);
    gemm(SXB, ap_wkT, Kb, nullptr, N_, TS_, TS_, 1, 0, 0, 0);
    gemm(SXB, ap_wvT, Vb, nullptr, N_, TS_, TS_, 1, 0, 0, 0);
    gemm(SXB, ap_wgT, Gb, nullptr, N_, TS_, TS_, 1, 0, 0, 0);
    k_ln<<<NN_, 128, 0, stream>>>(Z, P(blk, 36 + 6), P(blk, 36 + 7), nullptr, XB, TZ_, 0);
    gemm(XB, ap_wzT, G3, nullptr, NN_, 16, TZ_, 1, 0, 0, 0);  // bias [NN,16] (1x1 path)
    k_apb_attn<<<16, 192, 0, stream>>>(Qb, Kb, Vb, G3, maskv, Ob);
    k_gate_bf16<<<ew((size_t)N_ * TS_), 256, 0, stream>>>(SG16, Ob, Gb, (size_t)N_ * TS_);
    gemm(SG16, ap_woT, Pb, nullptr, N_, TS_, TS_, 1, 0, 0, 0);
    k_add<<<ew((size_t)N_ * TS_), 256, 0, stream>>>(S, Pb, (size_t)N_ * TS_);

    // --- sequence transition ---
    k_ln<<<N_, 128, 0, stream>>>(S, P(blk, 52 + 0), P(blk, 52 + 1), nullptr, SXB, TS_, 0);
    dual(SXB, ts_w1T, ts_w2T, SH16, N_, 1536, TS_);
    gemm(SH16, ts_w3T, Pb, nullptr, N_, TS_, 1536, 1, 0, 0, 0);
    k_add<<<ew((size_t)N_ * TS_), 256, 0, stream>>>(S, Pb, (size_t)N_ * TS_);
  }

  // outputs: s then z (tuple order)
  hipMemcpyAsync(d_out, S, (size_t)N_ * TS_ * 4, hipMemcpyDeviceToDevice, stream);
  hipMemcpyAsync((float*)d_out + (size_t)N_ * TS_, Z, (size_t)NN_ * TZ_ * 4,
                 hipMemcpyDeviceToDevice, stream);
}
